// GraphModelContrastivev5_20658792694716
// MI455X (gfx1250) — compile-verified, matrix operand order checked by reading, not run on hardware
//
#include <hip/hip_runtime.h>

#define NN   8192
#define EE   262144
#define MM   4096
#define SSZ  256
#define HTOK 768
#define DSM  512
#define DCO  251

typedef __attribute__((ext_vector_type(16))) _Float16 v16h;
typedef __attribute__((ext_vector_type(8)))  float    v8f;
typedef __attribute__((ext_vector_type(4)))  int      vi4;

union HVec { uint4 q[2]; v16h v; };

#if defined(__has_builtin)
#if __has_builtin(__builtin_amdgcn_global_load_async_to_lds_b128)
#define HAVE_ASYNC_LDS 1
#endif
#endif

__device__ __forceinline__ float lrelu(float x, float s) { return x >= 0.f ? x : s * x; }
__device__ __forceinline__ unsigned fkey(float f) {
  unsigned u = __float_as_uint(f);
  return (u & 0x80000000u) ? ~u : (u | 0x80000000u);
}
__device__ __forceinline__ float funkey(unsigned k) {
  unsigned u = (k & 0x80000000u) ? (k & 0x7FFFFFFFu) : ~k;
  return __uint_as_float(u);
}
__device__ __forceinline__ float wsum32(float v) {
#pragma unroll
  for (int o = 16; o > 0; o >>= 1) v += __shfl_xor(v, o, 32);
  return v;
}
__device__ __forceinline__ float wmax32(float v) {
#pragma unroll
  for (int o = 16; o > 0; o >>= 1) v = fmaxf(v, __shfl_xor(v, o, 32));
  return v;
}

// ---------------- conversions ----------------
__global__ void k_f32_to_f16(const float* __restrict__ in, _Float16* __restrict__ out, int n) {
  int i = blockIdx.x * blockDim.x + threadIdx.x;
  if (i < n) out[i] = (_Float16)in[i];
}

// out[n*K+k] = in[k*Nin+n] (f16), zero-padded for n in [Nin, Nout)
__global__ void k_transpose_f16(const float* __restrict__ in, _Float16* __restrict__ out,
                                int K, int Nin, int Nout) {
  int id = blockIdx.x * blockDim.x + threadIdx.x;
  if (id >= K * Nout) return;
  int k = id / Nout, n = id % Nout;
  float v = (n < Nin) ? in[(size_t)k * Nin + n] : 0.f;
  out[(size_t)n * K + k] = (_Float16)v;
}

// ---------------- WMMA GEMM: C[M,N] = A[M,K] * Bt[N,K]^T + bias ----------------
// Block = 8 waves = one 128(m) x 16(n) macro-tile. The 16-column B panel is staged
// into LDS (async DMA when available) and shared by all 8 waves; A streams from
// global as b128 pairs. K is paneled at KP halfs (32 KB LDS) for the K=2560 case.
// All klen values are multiples of 128, so the k-loop is unrolled 4x for pipelining.
#define KP 1024
__global__ void k_gemm_f16(const _Float16* __restrict__ A, const _Float16* __restrict__ Bt,
                           const float* __restrict__ bias, float* __restrict__ C,
                           int tilesN, int K, int lda, int ldb, int ldc,
                           int nBias, int mode) {
  __shared__ _Float16 lb[16 * KP];
  int wave = threadIdx.x >> 5, lane = threadIdx.x & 31;
  int bm = blockIdx.x / tilesN, tn = blockIdx.x % tilesN;
  int m0 = (bm * 8 + wave) * 16, n0 = tn * 16;
  int khalf = lane >> 4;
  int mrow = m0 + (lane & 15);
  int ncol = n0 + (lane & 15);
  const _Float16* ap  = A + (size_t)mrow * lda + khalf * 8;
  const _Float16* lbp = lb + (lane & 15) * KP + khalf * 8;
  v8f acc = {0.f, 0.f, 0.f, 0.f, 0.f, 0.f, 0.f, 0.f};

  for (int kp = 0; kp < K; kp += KP) {
    int klen = (K - kp < KP) ? (K - kp) : KP;
    int kc8 = klen >> 3;              // 16B chunks per column
    __syncthreads();                  // protect lb reuse across panels
    for (int c = threadIdx.x; c < 16 * kc8; c += 256) {
      int col = c / kc8;
      int ko = (c - col * kc8) * 8;
      const _Float16* g = Bt + (size_t)(n0 + col) * ldb + kp + ko;
      _Float16* l = lb + col * KP + ko;
#ifdef HAVE_ASYNC_LDS
      __builtin_amdgcn_global_load_async_to_lds_b128(
          (__attribute__((address_space(1))) vi4*)g,
          (__attribute__((address_space(3))) vi4*)l, 0, 0);
#else
      *(uint4*)l = *(const uint4*)g;
#endif
    }
#ifdef HAVE_ASYNC_LDS
    asm volatile("s_wait_asynccnt 0" ::: "memory");
#endif
    __syncthreads();
    const _Float16* app = ap + kp;
    for (int k0 = 0; k0 < klen; k0 += 128) {
#pragma unroll
      for (int u = 0; u < 4; u++) {
        HVec a, b;
        const uint4* aq = (const uint4*)(app + k0 + u * 32);
        const uint4* bq = (const uint4*)(lbp + k0 + u * 32);
        a.q[0] = aq[0]; a.q[1] = aq[2];   // K+0..7 and K+16..23 for this half-lane
        b.q[0] = bq[0]; b.q[1] = bq[2];
        acc = __builtin_amdgcn_wmma_f32_16x16x32_f16(false, a.v, false, b.v,
                                                     (short)0, acc, false, false);
      }
    }
  }

  float bv = (ncol < nBias) ? bias[ncol] : 0.f;
#pragma unroll
  for (int r = 0; r < 8; r++) {
    int row = m0 + khalf * 8 + r;
    float v = acc[r] + bv;
    if (mode == 1) v = lrelu(v, 0.01f);
    C[(size_t)row * ldc + ncol] = v;
  }
}

// ---------------- tiny 5x5 linear: out[n][j] = b[j] + sum_i x[n,i]*W[i*5+j] ----------------
__global__ void k_linear5(const float* __restrict__ x, int xstride,
                          const float* __restrict__ W, const float* __restrict__ b,
                          float* __restrict__ out, int n_nodes) {
  int n = blockIdx.x * blockDim.x + threadIdx.x;
  if (n >= n_nodes) return;
  float xi[5];
#pragma unroll
  for (int i = 0; i < 5; i++) xi[i] = x[(size_t)n * xstride + i];
#pragma unroll
  for (int j = 0; j < 5; j++) {
    float s = b[j];
#pragma unroll
    for (int i = 0; i < 5; i++) s += xi[i] * W[i * 5 + j];
    out[(size_t)n * 8 + j] = s;
  }
}

// ---------------- GATv2 ----------------
__global__ void k_gat_init(unsigned* __restrict__ smaxk, float* __restrict__ agg, int n_nodes) {
  int n = blockIdx.x * blockDim.x + threadIdx.x;
  if (n >= n_nodes) return;
  smaxk[n] = fkey(-__builtin_inff());
#pragma unroll
  for (int j = 0; j < 8; j++) agg[(size_t)n * 8 + j] = 0.f;
}

__global__ void k_gat_score(const int* __restrict__ src, const int* __restrict__ dst,
                            const float* __restrict__ xl, const float* __restrict__ xr,
                            const float* __restrict__ att, float* __restrict__ score,
                            unsigned* __restrict__ smaxk, int n_edges) {
  int e = blockIdx.x * blockDim.x + threadIdx.x;
  if (e >= n_edges) return;
  int s = src[e], d = dst[e];
  float sc = 0.f;
#pragma unroll
  for (int j = 0; j < 5; j++) {
    float m = xl[(size_t)s * 8 + j] + xr[(size_t)d * 8 + j];
    sc += lrelu(m, 0.2f) * att[j];
  }
  score[e] = sc;
  atomicMax(&smaxk[d], fkey(sc));
}

__global__ void k_gat_smax(const unsigned* __restrict__ smaxk, float* __restrict__ smaxf, int n_nodes) {
  int n = blockIdx.x * blockDim.x + threadIdx.x;
  if (n >= n_nodes) return;
  float f = funkey(smaxk[n]);
  smaxf[n] = __builtin_isfinite(f) ? f : 0.f;
}

__global__ void k_gat_accum(const int* __restrict__ src, const int* __restrict__ dst,
                            const float* __restrict__ xl, const float* __restrict__ score,
                            const float* __restrict__ smaxf, float* __restrict__ agg, int n_edges) {
  int e = blockIdx.x * blockDim.x + threadIdx.x;
  if (e >= n_edges) return;
  int s = src[e], d = dst[e];
  float ex = __expf(score[e] - smaxf[d]);
  atomicAdd(&agg[(size_t)d * 8 + 5], ex);
#pragma unroll
  for (int j = 0; j < 5; j++) atomicAdd(&agg[(size_t)d * 8 + j], ex * xl[(size_t)s * 8 + j]);
}

__global__ void k_gat_final(const float* __restrict__ agg, const float* __restrict__ bias,
                            float* __restrict__ out, int n_nodes) {
  int n = blockIdx.x * blockDim.x + threadIdx.x;
  if (n >= n_nodes) return;
  float inv = 1.f / fmaxf(agg[(size_t)n * 8 + 5], 1e-16f);
#pragma unroll
  for (int j = 0; j < 5; j++)
    out[(size_t)n * 8 + j] = lrelu(agg[(size_t)n * 8 + j] * inv + bias[j], 0.01f);
}

// ---------------- flash attention over all N (d=5), K/V tiles staged in LDS ----------------
__global__ void k_attn(const float* __restrict__ q8, const float* __restrict__ k8,
                       const float* __restrict__ v8, float* __restrict__ o8, int n_nodes) {
  __shared__ float sk[256 * 8];
  __shared__ float sv[256 * 8];
  int n = blockIdx.x * 256 + threadIdx.x;
  float qv[5];
#pragma unroll
  for (int j = 0; j < 5; j++) qv[j] = q8[(size_t)n * 8 + j];
  float mx = -3.0e38f, sum = 0.f, acc[5] = {0.f, 0.f, 0.f, 0.f, 0.f};
  const float scale = 0.4472135954999579f;  // 1/sqrt(5)
  for (int t0 = 0; t0 < n_nodes; t0 += 256) {
    __syncthreads();
    const uint4* kp = (const uint4*)(k8 + (size_t)(t0 + threadIdx.x) * 8);
    const uint4* vp = (const uint4*)(v8 + (size_t)(t0 + threadIdx.x) * 8);
    ((uint4*)sk)[threadIdx.x * 2 + 0] = kp[0];
    ((uint4*)sk)[threadIdx.x * 2 + 1] = kp[1];
    ((uint4*)sv)[threadIdx.x * 2 + 0] = vp[0];
    ((uint4*)sv)[threadIdx.x * 2 + 1] = vp[1];
    __syncthreads();
    for (int jj = 0; jj < 256; jj++) {
      const float* kr = sk + jj * 8;
      float d = qv[0]*kr[0] + qv[1]*kr[1] + qv[2]*kr[2] + qv[3]*kr[3] + qv[4]*kr[4];
      d *= scale;
      float nm = fmaxf(mx, d);
      float c = __expf(mx - nm), p = __expf(d - nm);
      sum = sum * c + p;
      const float* vr = sv + jj * 8;
#pragma unroll
      for (int j = 0; j < 5; j++) acc[j] = acc[j] * c + p * vr[j];
      mx = nm;
    }
  }
  float inv = 1.f / sum;
#pragma unroll
  for (int j = 0; j < 5; j++) o8[(size_t)n * 8 + j] = acc[j] * inv;
}

// ---------------- transformer epilogue: Wo + LN1 + FF(gelu) + LN2 ----------------
__global__ void k_tr_post(const float* __restrict__ x2, const float* __restrict__ o5,
                          const float* __restrict__ Wo, const float* __restrict__ bo,
                          const float* __restrict__ ln1g, const float* __restrict__ ln1b,
                          const float* __restrict__ W1, const float* __restrict__ b1,
                          const float* __restrict__ W2, const float* __restrict__ b2,
                          const float* __restrict__ ln2g, const float* __restrict__ ln2b,
                          float* __restrict__ xst, int n_nodes) {
  int n = blockIdx.x * blockDim.x + threadIdx.x;
  if (n >= n_nodes) return;
  float r[5];
#pragma unroll
  for (int j = 0; j < 5; j++) {
    float y = bo[j];
#pragma unroll
    for (int i = 0; i < 5; i++) y += o5[(size_t)n * 8 + i] * Wo[i * 5 + j];
    r[j] = x2[(size_t)n * 8 + j] + y;
  }
  float mu = 0.f;
#pragma unroll
  for (int j = 0; j < 5; j++) mu += r[j];
  mu *= 0.2f;
  float var = 0.f;
#pragma unroll
  for (int j = 0; j < 5; j++) { float d = r[j] - mu; var += d * d; }
  var *= 0.2f;
  float rs = rsqrtf(var + 1e-5f);
  float h[5];
#pragma unroll
  for (int j = 0; j < 5; j++) h[j] = (r[j] - mu) * rs * ln1g[j] + ln1b[j];
  float t1[10];
#pragma unroll
  for (int p = 0; p < 10; p++) {
    float t = b1[p];
#pragma unroll
    for (int j = 0; j < 5; j++) t += h[j] * W1[j * 10 + p];
    t1[p] = 0.5f * t * (1.f + erff(t * 0.7071067811865475f));
  }
  float r2[5];
#pragma unroll
  for (int j = 0; j < 5; j++) {
    float f = b2[j];
#pragma unroll
    for (int p = 0; p < 10; p++) f += t1[p] * W2[p * 5 + j];
    r2[j] = h[j] + f;
  }
  mu = 0.f;
#pragma unroll
  for (int j = 0; j < 5; j++) mu += r2[j];
  mu *= 0.2f;
  var = 0.f;
#pragma unroll
  for (int j = 0; j < 5; j++) { float d = r2[j] - mu; var += d * d; }
  var *= 0.2f;
  rs = rsqrtf(var + 1e-5f);
#pragma unroll
  for (int j = 0; j < 5; j++) xst[(size_t)n * 8 + j] = (r2[j] - mu) * rs * ln2g[j] + ln2b[j];
}

// ---------------- A_co[n, i*5+j] = x_sm[n,i] * x_st[n,j] (f16) ----------------
__global__ void k_outer(const float* __restrict__ xsm, const float* __restrict__ xst,
                        _Float16* __restrict__ A, int n_nodes) {
  int id = blockIdx.x * blockDim.x + threadIdx.x;
  if (id >= n_nodes * DSM) return;
  int n = id / DSM, i = id % DSM;
  float s = xsm[(size_t)n * DSM + i];
#pragma unroll
  for (int j = 0; j < 5; j++)
    A[(size_t)n * 2560 + i * 5 + j] = (_Float16)(s * xst[(size_t)n * 8 + j]);
}

// ---------------- fusion: 3 LayerNorms + avg/max + tiny MLPs + scaled concat (one wave / node) --
__global__ void k_fuse(const float* __restrict__ xsm, const float* __restrict__ xst,
                       const float* __restrict__ xco,
                       const float* __restrict__ smg, const float* __restrict__ smb,
                       const float* __restrict__ stg, const float* __restrict__ stb,
                       const float* __restrict__ cog, const float* __restrict__ cob,
                       const float* __restrict__ aW1, const float* __restrict__ ab1,
                       const float* __restrict__ aW2, const float* __restrict__ ab2,
                       const float* __restrict__ mW1, const float* __restrict__ mb1,
                       const float* __restrict__ mW2, const float* __restrict__ mb2,
                       _Float16* __restrict__ xfuse, int n_nodes) {
  int wave = threadIdx.x >> 5, lane = threadIdx.x & 31;
  int n = blockIdx.x * 8 + wave;
  if (n >= n_nodes) return;
  // ---- x_sm LN (512) ----
  const float* srow = xsm + (size_t)n * DSM;
  float vals[16], s = 0.f, s2 = 0.f;
#pragma unroll
  for (int k = 0; k < 16; k++) { float v = srow[lane + 32 * k]; vals[k] = v; s += v; s2 += v * v; }
  s = wsum32(s); s2 = wsum32(s2);
  float mu_sm = s / 512.f, var_sm = s2 / 512.f - mu_sm * mu_sm;
  float rs_sm = rsqrtf(fmaxf(var_sm, 0.f) + 1e-5f);
  float asum = 0.f, amax = -3e38f;
#pragma unroll
  for (int k = 0; k < 16; k++) {
    int idx = lane + 32 * k;
    float yn = (vals[k] - mu_sm) * rs_sm * smg[idx] + smb[idx];
    asum += yn; amax = fmaxf(amax, yn);
  }
  float sm_avg = wsum32(asum) / 512.f, sm_max = wmax32(amax);
  // ---- x_st LN (5), redundant per lane ----
  float stv[5];
#pragma unroll
  for (int j = 0; j < 5; j++) stv[j] = xst[(size_t)n * 8 + j];
  float mu_st = (stv[0] + stv[1] + stv[2] + stv[3] + stv[4]) * 0.2f;
  float var_st = 0.f;
#pragma unroll
  for (int j = 0; j < 5; j++) { float d = stv[j] - mu_st; var_st += d * d; }
  var_st *= 0.2f;
  float rs_st = rsqrtf(var_st + 1e-5f);
  float st_avg = 0.f, st_max = -3e38f;
#pragma unroll
  for (int j = 0; j < 5; j++) {
    float yn = (stv[j] - mu_st) * rs_st * stg[j] + stb[j];
    st_avg += yn; st_max = fmaxf(st_max, yn);
  }
  st_avg *= 0.2f;
  // ---- x_co LN (251) ----
  const float* crow = xco + (size_t)n * 256;
  float cv[8]; s = 0.f; s2 = 0.f;
#pragma unroll
  for (int k = 0; k < 8; k++) {
    int idx = lane + 32 * k;
    float v = (idx < DCO) ? crow[idx] : 0.f;
    cv[k] = v;
    if (idx < DCO) { s += v; s2 += v * v; }
  }
  s = wsum32(s); s2 = wsum32(s2);
  float mu_co = s / 251.f, var_co = s2 / 251.f - mu_co * mu_co;
  float rs_co = rsqrtf(fmaxf(var_co, 0.f) + 1e-5f);
  float csum = 0.f, cmax = -3e38f;
#pragma unroll
  for (int k = 0; k < 8; k++) {
    int idx = lane + 32 * k;
    if (idx < DCO) {
      float yn = (cv[k] - mu_co) * rs_co * cog[idx] + cob[idx];
      csum += yn; cmax = fmaxf(cmax, yn);
    }
  }
  float co_avg = wsum32(csum) / 251.f, co_max = wmax32(cmax);
  // ---- tiny MLPs (3->6->3) x2 ----
  float av[3] = {sm_avg, st_avg, co_avg};
  float mv[3] = {sm_max, st_max, co_max};
  float ha[6], hm[6];
#pragma unroll
  for (int p = 0; p < 6; p++) {
    float ta = ab1[p], tm = mb1[p];
#pragma unroll
    for (int c = 0; c < 3; c++) { ta += av[c] * aW1[c * 6 + p]; tm += mv[c] * mW1[c * 6 + p]; }
    ha[p] = lrelu(ta, 0.01f); hm[p] = lrelu(tm, 0.01f);
  }
  float cw[3];
#pragma unroll
  for (int c = 0; c < 3; c++) {
    float ta = ab2[c], tm = mb2[c];
#pragma unroll
    for (int p = 0; p < 6; p++) { ta += ha[p] * aW2[p * 3 + c]; tm += hm[p] * mW2[p * 3 + c]; }
    cw[c] = ta + tm;
  }
  // ---- write scaled concat [512 | 5 | 251] as f16 ----
  _Float16* orow = xfuse + (size_t)n * HTOK;
#pragma unroll
  for (int k = 0; k < 16; k++) {
    int idx = lane + 32 * k;
    float yn = (vals[k] - mu_sm) * rs_sm * smg[idx] + smb[idx];
    orow[idx] = (_Float16)(yn * cw[0]);
  }
  if (lane < 5) {
    float yn = (stv[lane] - mu_st) * rs_st * stg[lane] + stb[lane];
    orow[512 + lane] = (_Float16)(yn * cw[1]);
  }
#pragma unroll
  for (int k = 0; k < 8; k++) {
    int idx = lane + 32 * k;
    if (idx < DCO) {
      float yn = (cv[k] - mu_co) * rs_co * cog[idx] + cob[idx];
      orow[517 + idx] = (_Float16)(yn * cw[2]);
    }
  }
}

// ---------------- output stage ----------------
__global__ void k_gather(const float* __restrict__ xf, const int* __restrict__ idxs,
                         float* __restrict__ out, int rows) {
  int id = blockIdx.x * blockDim.x + threadIdx.x;
  if (id >= rows * 384) return;
  int m = id / 384, c = id % 384;
  out[id] = xf[(size_t)idxs[m] * 384 + c];
}

__global__ void k_sims(const float* __restrict__ target, const float* __restrict__ source,
                       const int* __restrict__ types, float* __restrict__ sims_out,
                       float* __restrict__ logits, int M) {
  int wave = threadIdx.x >> 5, lane = threadIdx.x & 31;
  int m = blockIdx.x * 8 + wave;
  if (m >= M) return;
  const float* t = target + (size_t)m * 384;
  const float* s = source + (size_t)types[m] * 384;
  float d = 0.f;
#pragma unroll
  for (int k = 0; k < 12; k++) { int c = lane + 32 * k; d += t[c] * s[c]; }
  d = wsum32(d);
  if (lane == 0) { sims_out[m] = d; logits[m] = d * 10.0f; }  // 1/TEMP
}

__global__ void k_zero_acc(float* __restrict__ acc) {
  if (threadIdx.x < 2) acc[threadIdx.x] = 0.f;
}

__global__ void k_loss_type(const float* __restrict__ logits, const int* __restrict__ types,
                            const int* __restrict__ labels, float* __restrict__ acc, int M) {
  int wave = threadIdx.x >> 5, lane = threadIdx.x & 31;
  int sId = blockIdx.x * 8 + wave;
  if (sId >= SSZ) return;
  float mA = -1e30f, mP = -1e30f;
  for (int m = lane; m < M; m += 32) {
    if (types[m] == sId) {
      float l = logits[m];
      mA = fmaxf(mA, l);
      if (labels[m] == 1) mP = fmaxf(mP, l);
    }
  }
  mA = wmax32(mA); mP = wmax32(mP);
  float sA = 0.f, sP = 0.f;
  for (int m = lane; m < M; m += 32) {
    if (types[m] == sId) {
      float l = logits[m];
      sA += __expf(l - mA);
      if (labels[m] == 1) sP += __expf(l - mP);
    }
  }
  sA = wsum32(sA); sP = wsum32(sP);
  if (lane == 0 && sP > 0.f) {
    float mlA = mA + logf(fmaxf(sA, 1e-30f));
    float mlP = mP + logf(fmaxf(sP, 1e-30f));
    atomicAdd(&acc[0], mlA - mlP);
    atomicAdd(&acc[1], 1.0f);
  }
}

__global__ void k_loss_fin(const float* __restrict__ acc, float* __restrict__ out) {
  if (threadIdx.x == 0) out[0] = acc[0] / fmaxf(acc[1], 1.0f);
}

// =====================================================================================
extern "C" void kernel_launch(void* const* d_in, const int* in_sizes, int n_in,
                              void* d_out, int out_size, void* d_ws, size_t ws_size,
                              hipStream_t stream) {
  (void)in_sizes; (void)n_in; (void)out_size; (void)ws_size;
  const float* node_emb = (const float*)d_in[1];
  const int*   edge     = (const int*)d_in[2];
  const int*   np_idx   = (const int*)d_in[3];
  const int*   np_type  = (const int*)d_in[4];
  const int*   np_label = (const int*)d_in[5];
  const int*   ch_idx   = (const int*)d_in[6];
  const float* st_emb   = (const float*)d_in[7];
  const float* P[64];
  for (int i = 8; i < 56; i++) P[i] = (const float*)d_in[i];

  const int* e_src = edge;
  const int* e_dst = edge + EE;

  // ---- workspace layout ----
  char* base = (char*)d_ws;
  size_t off = 0;
  auto alloc = [&](size_t bytes) { size_t o = off; off = (off + bytes + 255) & ~(size_t)255; return o; };
  _Float16* ne_h   = (_Float16*)(base + alloc((size_t)NN * HTOK * 2));
  _Float16* WsmT   = (_Float16*)(base + alloc((size_t)DSM * HTOK * 2));
  _Float16* WcoT   = (_Float16*)(base + alloc((size_t)256 * 2560 * 2));
  _Float16* WfT    = (_Float16*)(base + alloc((size_t)384 * HTOK * 2));
  float*    XL     = (float*)(base + alloc((size_t)NN * 8 * 4));
  float*    XR     = (float*)(base + alloc((size_t)NN * 8 * 4));
  float*    SCORE  = (float*)(base + alloc((size_t)EE * 4));
  unsigned* SMAXK  = (unsigned*)(base + alloc((size_t)NN * 4));
  float*    SMAXF  = (float*)(base + alloc((size_t)NN * 4));
  float*    AGG    = (float*)(base + alloc((size_t)NN * 8 * 4));
  float*    X1     = (float*)(base + alloc((size_t)NN * 8 * 4));
  float*    X2     = (float*)(base + alloc((size_t)NN * 8 * 4));
  float*    Q8     = (float*)(base + alloc((size_t)NN * 8 * 4));
  float*    K8     = (float*)(base + alloc((size_t)NN * 8 * 4));
  float*    V8     = (float*)(base + alloc((size_t)NN * 8 * 4));
  float*    O8     = (float*)(base + alloc((size_t)NN * 8 * 4));
  float*    XST    = (float*)(base + alloc((size_t)NN * 8 * 4));
  float*    XSM    = (float*)(base + alloc((size_t)NN * DSM * 4));
  _Float16* ACO    = (_Float16*)(base + alloc((size_t)NN * 2560 * 2));
  float*    XCO    = (float*)(base + alloc((size_t)NN * 256 * 4));
  _Float16* XFH    = (_Float16*)(base + alloc((size_t)NN * HTOK * 2));
  float*    XF     = (float*)(base + alloc((size_t)NN * 384 * 4));
  float*    LOGITS = (float*)(base + alloc((size_t)MM * 4));
  float*    ACC    = (float*)(base + alloc(8));

  float* out        = (float*)d_out;
  float* out_loss   = out;                 // [1]
  float* out_sims   = out + 1;             // [4096]
  float* out_target = out + 1 + MM;        // [4096,384]
  float* out_source = out_target + (size_t)MM * 384;  // [256,384]

  // ---- weight conversion / transposition ----
  k_f32_to_f16<<<(NN * HTOK + 255) / 256, 256, 0, stream>>>(node_emb, ne_h, NN * HTOK);
  k_transpose_f16<<<(HTOK * DSM + 255) / 256, 256, 0, stream>>>(P[36], WsmT, HTOK, DSM, DSM);
  k_transpose_f16<<<(2560 * 256 + 255) / 256, 256, 0, stream>>>(P[38], WcoT, 2560, DCO, 256);
  k_transpose_f16<<<(HTOK * 384 + 255) / 256, 256, 0, stream>>>(P[54], WfT, HTOK, 384, 384);

  // ---- GAT layers ----
  auto run_gat = [&](const float* xin, int xstride, const float* Wl, const float* bl,
                     const float* Wr, const float* br, const float* att, const float* bias,
                     float* xout) {
    k_linear5<<<NN / 256, 256, 0, stream>>>(xin, xstride, Wl, bl, XL, NN);
    k_linear5<<<NN / 256, 256, 0, stream>>>(xin, xstride, Wr, br, XR, NN);
    k_gat_init<<<NN / 256, 256, 0, stream>>>(SMAXK, AGG, NN);
    k_gat_score<<<EE / 256, 256, 0, stream>>>(e_src, e_dst, XL, XR, att, SCORE, SMAXK, EE);
    k_gat_smax<<<NN / 256, 256, 0, stream>>>(SMAXK, SMAXF, NN);
    k_gat_accum<<<EE / 256, 256, 0, stream>>>(e_src, e_dst, XL, SCORE, SMAXF, AGG, EE);
    k_gat_final<<<NN / 256, 256, 0, stream>>>(AGG, bias, xout, NN);
  };
  run_gat(st_emb, 5, P[8], P[9], P[10], P[11], P[12], P[13], X1);
  run_gat(X1,    8, P[14], P[15], P[16], P[17], P[18], P[19], X2);

  // ---- transformer ----
  k_linear5<<<NN / 256, 256, 0, stream>>>(X2, 8, P[20], P[21], Q8, NN);
  k_linear5<<<NN / 256, 256, 0, stream>>>(X2, 8, P[22], P[23], K8, NN);
  k_linear5<<<NN / 256, 256, 0, stream>>>(X2, 8, P[24], P[25], V8, NN);
  k_attn<<<NN / 256, 256, 0, stream>>>(Q8, K8, V8, O8, NN);
  k_tr_post<<<NN / 256, 256, 0, stream>>>(X2, O8, P[26], P[27], P[28], P[29],
                                          P[30], P[31], P[32], P[33], P[34], P[35], XST, NN);

  // ---- x_sm GEMM: [8192,768]x[768,512] ----
  k_gemm_f16<<<(NN / 128) * (DSM / 16), 256, 0, stream>>>(
      ne_h, WsmT, P[37], XSM, DSM / 16, HTOK, HTOK, HTOK, DSM, DSM, 0);

  // ---- outer product + x_co GEMM: [8192,2560]x[2560,251] ----
  k_outer<<<(NN * DSM + 255) / 256, 256, 0, stream>>>(XSM, XST, ACO, NN);
  k_gemm_f16<<<(NN / 128) * (256 / 16), 256, 0, stream>>>(
      ACO, WcoT, P[39], XCO, 256 / 16, 2560, 2560, 2560, 256, DCO, 0);

  // ---- fusion (LNs + attention-weights MLPs + scaled concat) ----
  k_fuse<<<NN / 8, 256, 0, stream>>>(XSM, XST, XCO,
                                     P[40], P[41], P[42], P[43], P[44], P[45],
                                     P[46], P[47], P[48], P[49], P[50], P[51], P[52], P[53],
                                     XFH, NN);

  // ---- Wf GEMM + leaky: [8192,768]x[768,384] ----
  k_gemm_f16<<<(NN / 128) * (384 / 16), 256, 0, stream>>>(
      XFH, WfT, P[55], XF, 384 / 16, HTOK, HTOK, HTOK, 384, 384, 1);

  // ---- outputs: target/source gathers, sims, InfoNCE ----
  k_gather<<<(MM * 384 + 255) / 256, 256, 0, stream>>>(XF, np_idx, out_target, MM);
  k_gather<<<(SSZ * 384 + 255) / 256, 256, 0, stream>>>(XF, ch_idx, out_source, SSZ);
  k_sims<<<MM / 8, 256, 0, stream>>>(out_target, out_source, np_type, out_sims, LOGITS, MM);
  k_zero_acc<<<1, 32, 0, stream>>>(ACC);
  k_loss_type<<<SSZ / 8, 256, 0, stream>>>(LOGITS, np_type, np_label, ACC, MM);
  k_loss_fin<<<1, 32, 0, stream>>>(ACC, out_loss);
}
